// MoHAttention_22445499089593
// MI455X (gfx1250) — matrix operand order repaired and checked by reading, not compiled, not run on hardware
//
#include <hip/hip_runtime.h>
#include <hip/hip_bf16.h>

// ---------------- problem constants ----------------
constexpr int kB = 32, kN = 577, kH = 12, kD = 64, kC = 768;
constexpr int kM = kB * kN;              // 18464 rows (multiple of 16)
constexpr int kNPAD = 592;               // 577 padded to multiple of 16
constexpr long long kSQ   = (long long)kB * kH * kNPAD * kD;  // q/k/v buf elems
constexpr long long kYE   = (long long)kM * kC;               // y_full elems
constexpr long long kAOFF = (long long)kM * kC;               // attn_full offset in d_out (floats)

// workspace layout (bytes, all 256B aligned)
constexpr size_t OFF_Q   = 0;
constexpr size_t OFF_K   = OFF_Q + (size_t)kSQ * 2;
constexpr size_t OFF_V   = OFF_K + (size_t)kSQ * 2;
constexpr size_t OFF_Y   = OFF_V + (size_t)kSQ * 2;
constexpr size_t OFF_SC  = OFF_Y + (size_t)kYE * 2;
constexpr size_t OFF_IDX = OFF_SC + 256;

// ---------------- WMMA helpers (CDNA5 wave32, bf16 16x16x32) ----------------
typedef __attribute__((ext_vector_type(16))) __bf16 v16bf;
typedef __attribute__((ext_vector_type(8)))  __bf16 v8bf;
typedef __attribute__((ext_vector_type(8)))  float  v8f;

// A-matrix 16x32 (MxK) fragment from an LDS tile with row stride `stride`,
// K window starting at k0. Lane l<16 row M=l: e0..7->K=e, e8..15->K=e+8;
// lane l>=16 row M=l-16: e0..7->K=e+8, e8..15->K=e+16.
__device__ __forceinline__ v16bf load_a_frag_s(const __bf16* As, int stride,
                                               int mBase, int k0, int lane) {
  int m  = mBase + (lane & 15);
  int hi = (lane >> 4) & 1;
  const __bf16* p = As + m * stride + k0 + hi * 8;
  v8bf lo = *(const v8bf*)p;
  v8bf hh = *(const v8bf*)(p + 16);
  v16bf a;
#pragma unroll
  for (int e = 0; e < 8; ++e) { a[e] = lo[e]; a[e + 8] = hh[e]; }
  return a;
}

// B-matrix 32x16 (KxN) fragment from LDS tile [k][n] with row stride `stride`.
// Lane l column N = nBase + l%16; half h: e0..7->K=8h+e, e8..15->K=16+8h+(e-8).
__device__ __forceinline__ v16bf load_b_frag_s(const __bf16* Bs, int stride,
                                               int k0, int nBase, int lane) {
  int n  = nBase + (lane & 15);
  int hi = (lane >> 4) & 1;
  const __bf16* p = Bs + (k0 + hi * 8) * stride + n;
  v16bf b;
#pragma unroll
  for (int e = 0; e < 8; ++e) { b[e] = p[e * stride]; b[e + 8] = p[(e + 16) * stride]; }
  return b;
}

#define WMMA_BF16(a, b, c) \
  __builtin_amdgcn_wmma_f32_16x16x32_bf16(false, (a), false, (b), (short)0, (c), false, false)

// ---------------- Kernel 1: qkv = x @ qkv_w^T + bias, scattered to bf16 q/k/v ----------------
__global__ __launch_bounds__(256) void qkv_gemm_kernel(
    const float* __restrict__ x, const float* __restrict__ w, const float* __restrict__ bias,
    __bf16* __restrict__ qb, __bf16* __restrict__ kb, __bf16* __restrict__ vb) {
  __shared__ __align__(16) __bf16 As[64 * 32];
  __shared__ __align__(16) __bf16 Bs[32 * 64];
  const int mBlk = blockIdx.y * 64, nBlk = blockIdx.x * 64;
  const int t = threadIdx.x, wave = t >> 5, lane = t & 31;
  const int wm = (wave >> 1) * 16, wn = (wave & 1) * 32;
  v8f c0 = {}, c1 = {};
  for (int k0 = 0; k0 < kC; k0 += 32) {
    for (int i = t; i < 64 * 32; i += 256) {
      int r = i >> 5, kk = i & 31, gr = mBlk + r;
      float v = (gr < kM) ? x[(size_t)gr * kC + k0 + kk] : 0.f;
      As[r * 32 + kk] = (__bf16)v;
    }
    for (int i = t; i < 64 * 32; i += 256) {
      int n = i >> 5, kk = i & 31;
      Bs[kk * 64 + n] = (__bf16)w[(size_t)(nBlk + n) * kC + k0 + kk];
    }
    __syncthreads();
    v16bf a  = load_a_frag_s(As, 32, wm, 0, lane);
    v16bf b0 = load_b_frag_s(Bs, 64, 0, wn, lane);
    v16bf b1 = load_b_frag_s(Bs, 64, 0, wn + 16, lane);
    c0 = WMMA_BF16(a, b0, c0);
    c1 = WMMA_BF16(a, b1, c1);
    __syncthreads();
  }
  const int nl = lane & 15, mAdd = (lane >> 4) * 8;
#pragma unroll
  for (int tn = 0; tn < 2; ++tn) {
    v8f cc = tn ? c1 : c0;
    int gc = nBlk + wn + tn * 16 + nl;
#pragma unroll
    for (int j = 0; j < 8; ++j) {
      int gr = mBlk + wm + j + mAdd;
      if (gr >= kM) continue;
      float val = cc[j] + bias[gc];
      int which = gc / kC, rem = gc % kC;
      int h = rem >> 6, d = rem & 63;
      int bb = gr / kN, nn = gr % kN;
      __bf16* dst = (which == 0) ? qb : (which == 1) ? kb : vb;
      dst[(((size_t)bb * kH + h) * kNPAD + nn) * kD + d] = (__bf16)val;
    }
  }
}

// ---------------- Kernel 2: per-head score = mean(q^2) ----------------
__global__ __launch_bounds__(256) void head_score_kernel(const __bf16* __restrict__ qb,
                                                         float* __restrict__ scores) {
  const int h = blockIdx.x;
  const int total = kB * kN * kD;
  float acc = 0.f;
  for (int i = threadIdx.x; i < total; i += 256) {
    int bb = i / (kN * kD);
    int rem = i % (kN * kD);
    int nn = rem >> 6, d = rem & 63;
    float v = (float)qb[(((size_t)bb * kH + h) * kNPAD + nn) * kD + d];
    acc += v * v;
  }
  __shared__ float red[256];
  red[threadIdx.x] = acc;
  __syncthreads();
  for (int s = 128; s > 0; s >>= 1) {
    if (threadIdx.x < s) red[threadIdx.x] += red[threadIdx.x + s];
    __syncthreads();
  }
  if (threadIdx.x == 0) scores[h] = red[0] / (float)total;
}

// ---------------- Kernel 3: top-4 (sorted) + 8 unselected head indices ----------------
__global__ void topk_kernel(const float* __restrict__ scores, int* __restrict__ idx) {
  if (threadIdx.x != 0) return;
  float s[kH]; bool used[kH]; int sel[4];
  for (int h = 0; h < kH; ++h) { s[h] = scores[h]; used[h] = false; }
  for (int t = 0; t < 4; ++t) {
    int bi = 0; float bv = -1e30f;
    for (int h = 0; h < kH; ++h)
      if (!used[h] && s[h] > bv) { bv = s[h]; bi = h; }
    used[bi] = true; sel[t] = bi;
  }
  for (int a = 0; a < 4; ++a)
    for (int b2 = a + 1; b2 < 4; ++b2)
      if (sel[b2] < sel[a]) { int tmp = sel[a]; sel[a] = sel[b2]; sel[b2] = tmp; }
  for (int t = 0; t < 4; ++t) idx[t] = sel[t];
  int cnt = 4;
  for (int h = 0; h < kH; ++h)
    if (!used[h]) idx[cnt++] = h;       // 8 unselected, ascending
}

// ---------------- Kernel 4: zero attn_full slices of the 8 unselected heads ----------------
// Write-once data far larger than L2 -> non-temporal scalar stores, lane-contiguous.
__global__ __launch_bounds__(256) void zero_unsel_kernel(const int* __restrict__ idx,
                                                         float* __restrict__ out) {
  const int hu = idx[4 + blockIdx.y];
  const int bb = blockIdx.z;
  float* base = out + kAOFF + (((size_t)bb * kH + hu) * kN) * kN;
  const long long total = (long long)kN * kN;  // 332929
  long long ibase = (long long)blockIdx.x * 1024 + threadIdx.x;
#pragma unroll
  for (int r = 0; r < 4; ++r) {
    long long i = ibase + (long long)r * 256;
    if (i < total) __builtin_nontemporal_store(0.f, base + i);
  }
}

// ---------------- Kernel 5: fused flash-style attention ----------------
// Per (b, selected head, 64-row block): 3 passes over K/V column tiles.
//   Pass A: row max of S = scale*QK^T    (WMMA + shfl reductions, no stores)
//   Pass B: row sum of exp(S - m)        (recompute S with WMMA)
//   Pass C: p = exp(S-m)/l -> attn_full (single NT write) + O += P@V (WMMA)
__global__ __launch_bounds__(256) void fused_attn_kernel(
    const __bf16* __restrict__ qb, const __bf16* __restrict__ kb,
    const __bf16* __restrict__ vb, const int* __restrict__ idx,
    float* __restrict__ out, __bf16* __restrict__ yb) {
  __shared__ __align__(16) __bf16 Qs[64 * 64];
  __shared__ __align__(16) __bf16 Ks[64 * 64];   // [d][n]
  __shared__ __align__(16) __bf16 Vs[64 * 64];   // [j][d]
  __shared__ __align__(16) __bf16 Ps[64 * 64];   // [m][j]
  __shared__ float partial[8][16];
  __shared__ float mrow[64], lrow[64], invl[64];

  const int z = blockIdx.z, bb = z >> 2, h = idx[z & 3];
  const int mBlk = blockIdx.y * 64;
  const int t = threadIdx.x, wave = t >> 5, lane = t & 31;
  const int wm = (wave >> 1) * 16, wn = (wave & 1) * 32;
  const int nl = lane & 15, mAdd = (lane >> 4) * 8;

  const __bf16* qrow = qb + ((size_t)bb * kH + h) * kNPAD * kD;
  const __bf16* krow = kb + ((size_t)bb * kH + h) * kNPAD * kD;
  const __bf16* vrow = vb + ((size_t)bb * kH + h) * kNPAD * kD;
  float* attn = out + kAOFF + (((size_t)bb * kH + h) * kN) * kN;

  // stage Q tile [m][d] once; zero padded rows
  for (int i = t; i < 64 * 64; i += 256) {
    int r = i >> 6, d = i & 63, gr = mBlk + r;
    Qs[i] = (gr < kN) ? qrow[(size_t)gr * kD + d] : (__bf16)0.f;
  }
  if (t < 64) { mrow[t] = -3.0e38f; lrow[t] = 0.f; }
  __syncthreads();

  // -------- Pass A: row max --------
  for (int jb = 0; jb < 10; ++jb) {
    const int jBlk = jb * 64;
    for (int i = t; i < 64 * 64; i += 256) {
      int d = i >> 6, n = i & 63, gn = jBlk + n;
      Ks[i] = (gn < kN) ? krow[(size_t)gn * kD + d] : (__bf16)0.f;
    }
    {  // prefetch next K tile (clamped; no divergence)
      int nj = jBlk + 64 + (t & 63);
      if (nj > kNPAD - 1) nj = kNPAD - 1;
      __builtin_prefetch(krow + (size_t)nj * kD, 0, 1);
    }
    __syncthreads();
    v8f s0 = {}, s1 = {};
#pragma unroll
    for (int k0 = 0; k0 < 64; k0 += 32) {
      v16bf a  = load_a_frag_s(Qs, 64, wm, k0, lane);
      v16bf b0 = load_b_frag_s(Ks, 64, k0, wn, lane);
      v16bf b1 = load_b_frag_s(Ks, 64, k0, wn + 16, lane);
      s0 = WMMA_BF16(a, b0, s0);
      s1 = WMMA_BF16(a, b1, s1);
    }
    const bool ok0 = (jBlk + wn + nl) < kN;
    const bool ok1 = (jBlk + wn + 16 + nl) < kN;
#pragma unroll
    for (int j = 0; j < 8; ++j) {
      float a0 = ok0 ? s0[j] * 0.125f : -3.0e38f;
      float a1 = ok1 ? s1[j] * 0.125f : -3.0e38f;
      float m = fmaxf(a0, a1);
#pragma unroll
      for (int msk = 1; msk < 16; msk <<= 1) m = fmaxf(m, __shfl_xor(m, msk, 32));
      if (nl == 0) partial[wave][j + mAdd] = m;
    }
    __syncthreads();
    if (t < 64) {
      float m = fmaxf(partial[(t >> 4) * 2][t & 15], partial[(t >> 4) * 2 + 1][t & 15]);
      mrow[t] = fmaxf(mrow[t], m);
    }
    __syncthreads();
  }

  // -------- Pass B: row sum of exp(S - m) --------
  for (int jb = 0; jb < 10; ++jb) {
    const int jBlk = jb * 64;
    for (int i = t; i < 64 * 64; i += 256) {
      int d = i >> 6, n = i & 63, gn = jBlk + n;
      Ks[i] = (gn < kN) ? krow[(size_t)gn * kD + d] : (__bf16)0.f;
    }
    __syncthreads();
    v8f s0 = {}, s1 = {};
#pragma unroll
    for (int k0 = 0; k0 < 64; k0 += 32) {
      v16bf a  = load_a_frag_s(Qs, 64, wm, k0, lane);
      v16bf b0 = load_b_frag_s(Ks, 64, k0, wn, lane);
      v16bf b1 = load_b_frag_s(Ks, 64, k0, wn + 16, lane);
      s0 = WMMA_BF16(a, b0, s0);
      s1 = WMMA_BF16(a, b1, s1);
    }
    const bool ok0 = (jBlk + wn + nl) < kN;
    const bool ok1 = (jBlk + wn + 16 + nl) < kN;
#pragma unroll
    for (int j = 0; j < 8; ++j) {
      float m = mrow[wm + j + mAdd];
      float a0 = ok0 ? expf(s0[j] * 0.125f - m) : 0.f;
      float a1 = ok1 ? expf(s1[j] * 0.125f - m) : 0.f;
      float sum = a0 + a1;
#pragma unroll
      for (int msk = 1; msk < 16; msk <<= 1) sum += __shfl_xor(sum, msk, 32);
      if (nl == 0) partial[wave][j + mAdd] = sum;
    }
    __syncthreads();
    if (t < 64) lrow[t] += partial[(t >> 4) * 2][t & 15] + partial[(t >> 4) * 2 + 1][t & 15];
    __syncthreads();
  }
  if (t < 64) invl[t] = 1.f / lrow[t];
  __syncthreads();

  // -------- Pass C: write p once (NT) + accumulate O = P @ V --------
  v8f o0 = {}, o1 = {};
  for (int jb = 0; jb < 10; ++jb) {
    const int jBlk = jb * 64;
    for (int i = t; i < 64 * 64; i += 256) {
      int d = i >> 6, n = i & 63, gn = jBlk + n;
      Ks[i] = (gn < kN) ? krow[(size_t)gn * kD + d] : (__bf16)0.f;
    }
    for (int i = t; i < 64 * 64; i += 256) {
      int r = i >> 6, d = i & 63, gj = jBlk + r;
      Vs[i] = (gj < kN) ? vrow[(size_t)gj * kD + d] : (__bf16)0.f;
    }
    {  // prefetch next K and V tiles (clamped; no divergence)
      int nj = jBlk + 64 + (t & 63);
      if (nj > kNPAD - 1) nj = kNPAD - 1;
      __builtin_prefetch(krow + (size_t)nj * kD, 0, 1);
      __builtin_prefetch(vrow + (size_t)nj * kD, 0, 1);
    }
    __syncthreads();
    v8f s0 = {}, s1 = {};
#pragma unroll
    for (int k0 = 0; k0 < 64; k0 += 32) {
      v16bf a  = load_a_frag_s(Qs, 64, wm, k0, lane);
      v16bf b0 = load_b_frag_s(Ks, 64, k0, wn, lane);
      v16bf b1 = load_b_frag_s(Ks, 64, k0, wn + 16, lane);
      s0 = WMMA_BF16(a, b0, s0);
      s1 = WMMA_BF16(a, b1, s1);
    }
    const bool ok0 = (jBlk + wn + nl) < kN;
    const bool ok1 = (jBlk + wn + 16 + nl) < kN;
#pragma unroll
    for (int j = 0; j < 8; ++j) {
      int r = wm + j + mAdd;
      float m = mrow[r], il = invl[r];
      float p0 = ok0 ? expf(s0[j] * 0.125f - m) * il : 0.f;
      float p1 = ok1 ? expf(s1[j] * 0.125f - m) * il : 0.f;
      Ps[r * 64 + wn + nl] = (__bf16)p0;
      Ps[r * 64 + wn + 16 + nl] = (__bf16)p1;
      int gr = mBlk + r;
      if (gr < kN) {
        if (ok0) __builtin_nontemporal_store(p0, &attn[(size_t)gr * kN + jBlk + wn + nl]);
        if (ok1) __builtin_nontemporal_store(p1, &attn[(size_t)gr * kN + jBlk + wn + 16 + nl]);
      }
    }
    __syncthreads();
#pragma unroll
    for (int k0 = 0; k0 < 64; k0 += 32) {
      v16bf a  = load_a_frag_s(Ps, 64, wm, k0, lane);
      v16bf b0 = load_b_frag_s(Vs, 64, k0, wn, lane);
      v16bf b1 = load_b_frag_s(Vs, 64, k0, wn + 16, lane);
      o0 = WMMA_BF16(a, b0, o0);
      o1 = WMMA_BF16(a, b1, o1);
    }
    __syncthreads();
  }

  // store O rows into bf16 y_full (head-scattered)
#pragma unroll
  for (int tn = 0; tn < 2; ++tn) {
    v8f cc = tn ? o1 : o0;
    int gc = wn + tn * 16 + nl;  // d index 0..63
#pragma unroll
    for (int j = 0; j < 8; ++j) {
      int gr = mBlk + wm + j + mAdd;
      if (gr < kN) yb[((size_t)bb * kN + gr) * kC + h * kD + gc] = (__bf16)cc[j];
    }
  }
}

// ---------------- Kernel 6: x_out = y_full @ proj_w^T + proj_b ----------------
__global__ __launch_bounds__(256) void proj_kernel(
    const __bf16* __restrict__ yb, const float* __restrict__ pw,
    const float* __restrict__ pb, float* __restrict__ out) {
  __shared__ __align__(16) __bf16 As[64 * 32];
  __shared__ __align__(16) __bf16 Bs[32 * 64];
  const int mBlk = blockIdx.y * 64, nBlk = blockIdx.x * 64;
  const int t = threadIdx.x, wave = t >> 5, lane = t & 31;
  const int wm = (wave >> 1) * 16, wn = (wave & 1) * 32;
  v8f c0 = {}, c1 = {};
  for (int k0 = 0; k0 < kC; k0 += 32) {
    for (int i = t; i < 64 * 32; i += 256) {
      int r = i >> 5, kk = i & 31, gr = mBlk + r;
      As[r * 32 + kk] = (gr < kM) ? yb[(size_t)gr * kC + k0 + kk] : (__bf16)0.f;
    }
    for (int i = t; i < 64 * 32; i += 256) {
      int n = i >> 5, kk = i & 31;
      Bs[kk * 64 + n] = (__bf16)pw[(size_t)(nBlk + n) * kC + k0 + kk];
    }
    __syncthreads();
    v16bf a  = load_a_frag_s(As, 32, wm, 0, lane);
    v16bf b0 = load_b_frag_s(Bs, 64, 0, wn, lane);
    v16bf b1 = load_b_frag_s(Bs, 64, 0, wn + 16, lane);
    c0 = WMMA_BF16(a, b0, c0);
    c1 = WMMA_BF16(a, b1, c1);
    __syncthreads();
  }
  const int nl = lane & 15, mAdd = (lane >> 4) * 8;
#pragma unroll
  for (int tn = 0; tn < 2; ++tn) {
    v8f cc = tn ? c1 : c0;
    int gc = nBlk + wn + tn * 16 + nl;
#pragma unroll
    for (int j = 0; j < 8; ++j) {
      int gr = mBlk + wm + j + mAdd;
      if (gr < kM) __builtin_nontemporal_store(cc[j] + pb[gc], &out[(size_t)gr * kC + gc]);
    }
  }
}

// ---------------- launch ----------------
extern "C" void kernel_launch(void* const* d_in, const int* in_sizes, int n_in,
                              void* d_out, int out_size, void* d_ws, size_t ws_size,
                              hipStream_t stream) {
  (void)in_sizes; (void)n_in; (void)ws_size; (void)out_size;
  const float* x      = (const float*)d_in[0];
  const float* qkv_w  = (const float*)d_in[1];
  const float* qkv_b  = (const float*)d_in[2];
  const float* proj_w = (const float*)d_in[3];
  const float* proj_b = (const float*)d_in[4];
  float* out = (float*)d_out;
  char* ws = (char*)d_ws;
  __bf16* qb = (__bf16*)(ws + OFF_Q);
  __bf16* kb = (__bf16*)(ws + OFF_K);
  __bf16* vb = (__bf16*)(ws + OFF_V);
  __bf16* yb = (__bf16*)(ws + OFF_Y);
  float* scores = (float*)(ws + OFF_SC);
  int* idx = (int*)(ws + OFF_IDX);

  // only y_full needs a blanket zero (28 MB); attn_full zeros are selective,
  // x_out is fully overwritten by proj_kernel.
  hipMemsetAsync(ws + OFF_Y, 0, (size_t)kYE * 2, stream);

  qkv_gemm_kernel<<<dim3(2304 / 64, (kM + 63) / 64), 256, 0, stream>>>(x, qkv_w, qkv_b, qb, kb, vb);
  head_score_kernel<<<kH, 256, 0, stream>>>(qb, scores);
  topk_kernel<<<1, 32, 0, stream>>>(scores, idx);
  zero_unsel_kernel<<<dim3((kN * kN + 1023) / 1024, kH - 4, kB), 256, 0, stream>>>(idx, out);
  fused_attn_kernel<<<dim3(1, 10, kB * 4), 256, 0, stream>>>(qb, kb, vb, idx, out, yb);
  proj_kernel<<<dim3(kC / 64, (kM + 63) / 64), 256, 0, stream>>>(yb, proj_w, proj_b, out);
}
